// TempGAT_9345848836493
// MI455X (gfx1250) — matrix operand order repaired and checked by reading, not compiled, not run on hardware
//
#include <hip/hip_runtime.h>

// ---------------------------------------------------------------------------
// GAT on MI455X (gfx1250). Wave32 WMMA f32_16x16x32_f16 everywhere.
// Pipeline:
//   1) convert features -> f16, W_heads/W_out -> f16 transposed
//   2) h1 = x @ W  (WMMA), also emit h1^T in f16 for the P@V stage
//   3) s1/s2 scores (wave reductions)
//   4) softmax stats per row (max, 1/Z) over masked leaky-relu scores
//   5) fused attention apply: regenerate P tiles in registers -> WMMA P@V,
//      fused ELU, pack [N, H*64] f16
//   6) layer 2: same GEMM/scores/stats/apply, final f32 -> d_out
// ---------------------------------------------------------------------------

typedef __attribute__((ext_vector_type(16))) _Float16 v16h;
typedef __attribute__((ext_vector_type(8)))  _Float16 v8h;
typedef __attribute__((ext_vector_type(8)))  float    v8f;

#define N_NODES 4096
#define NEG_INF_F (-9000000000000000.0f)
#define LRELU_ALPHA 0.2f

// ---------------------------------------------------------------- converts
__global__ void k_f32_to_f16(const float* __restrict__ src,
                             _Float16* __restrict__ dst, int n) {
  int i = blockIdx.x * blockDim.x + threadIdx.x;
  if (i < n) dst[i] = (_Float16)src[i];
}

// src: [H][I][D] f32 -> dst: [H][D][I] f16   (flat total = H*I*D)
__global__ void k_transpose_f16(const float* __restrict__ src,
                                _Float16* __restrict__ dst,
                                int I, int D, int total) {
  int idx = blockIdx.x * blockDim.x + threadIdx.x;
  if (idx >= total) return;
  int h   = idx / (I * D);
  int rem = idx - h * (I * D);
  int i   = rem / D;
  int d   = rem - i * D;
  dst[((size_t)h * D + d) * I + i] = (_Float16)src[idx];
}

// ---------------------------------------------------------------- GEMM 16x64
// out[h][rowbase..rowbase+15][0..63] = A[rows][K] * B[h][K][64]
// A  row-major f16 [N][K];  BT f16 [H][64][K] (i.e. B transposed)
// One wave per 16x64 tile; 4 WMMA accumulators; K stepped by 32.
__global__ __launch_bounds__(32)
void k_gemm16(const _Float16* __restrict__ A,
              const _Float16* __restrict__ BT,
              int K,
              float* __restrict__ out,        // [H][N][64]
              _Float16* __restrict__ outT) {  // [H][64][N]
  const int h       = blockIdx.y;
  const int rowbase = blockIdx.x * 16;
  const int lane    = threadIdx.x;
  const int ml      = lane & 15;
  const int hi      = lane >> 4;

  const _Float16* Arow = A + (size_t)(rowbase + ml) * K;
  const _Float16* Bh   = BT + (size_t)h * 64 * K;

  v8f c0 = {}, c1 = {}, c2 = {}, c3 = {};
  for (int kt = 0; kt < K; kt += 32) {
    v8h alo = *(const v8h*)(Arow + kt + hi * 16);
    v8h ahi = *(const v8h*)(Arow + kt + hi * 16 + 8);
    v16h a;
#pragma unroll
    for (int s = 0; s < 8; ++s) { a[s] = alo[s]; a[s + 8] = ahi[s]; }

    v16h b0 = *(const v16h*)(Bh + (size_t)(0  + ml) * K + kt + hi * 16);
    v16h b1 = *(const v16h*)(Bh + (size_t)(16 + ml) * K + kt + hi * 16);
    v16h b2 = *(const v16h*)(Bh + (size_t)(32 + ml) * K + kt + hi * 16);
    v16h b3 = *(const v16h*)(Bh + (size_t)(48 + ml) * K + kt + hi * 16);

    c0 = __builtin_amdgcn_wmma_f32_16x16x32_f16(false, a, false, b0, (short)0, c0, false, false);
    c1 = __builtin_amdgcn_wmma_f32_16x16x32_f16(false, a, false, b1, (short)0, c1, false, false);
    c2 = __builtin_amdgcn_wmma_f32_16x16x32_f16(false, a, false, b2, (short)0, c2, false, false);
    c3 = __builtin_amdgcn_wmma_f32_16x16x32_f16(false, a, false, b3, (short)0, c3, false, false);
  }

  float*     outh  = out  + ((size_t)h * N_NODES + rowbase) * 64;
  _Float16*  outTh = outT + (size_t)h * 64 * N_NODES;
#pragma unroll
  for (int i = 0; i < 8; ++i) {
    int gr   = i + hi * 8;          // row within tile
    int grow = rowbase + gr;        // global row
    float v0 = c0[i], v1 = c1[i], v2 = c2[i], v3 = c3[i];
    outh[(size_t)gr * 64 +  0 + ml] = v0;
    outh[(size_t)gr * 64 + 16 + ml] = v1;
    outh[(size_t)gr * 64 + 32 + ml] = v2;
    outh[(size_t)gr * 64 + 48 + ml] = v3;
    outTh[(size_t)( 0 + ml) * N_NODES + grow] = (_Float16)v0;
    outTh[(size_t)(16 + ml) * N_NODES + grow] = (_Float16)v1;
    outTh[(size_t)(32 + ml) * N_NODES + grow] = (_Float16)v2;
    outTh[(size_t)(48 + ml) * N_NODES + grow] = (_Float16)v3;
  }
}

// ------------------------------------------------------------- score vectors
// s1[h][n] = h[h][n][:] . a[h][0:64],  s2 with a[h][64:128]. One wave per (n,h).
__global__ __launch_bounds__(32)
void k_scores(const float* __restrict__ Hmat,   // [H][N][64]
              const float* __restrict__ a,      // [H][128]
              float* __restrict__ s1, float* __restrict__ s2) {
  int n = blockIdx.x, h = blockIdx.y, lane = threadIdx.x;
  const float* row = Hmat + ((size_t)h * N_NODES + n) * 64;
  const float* ah  = a + (size_t)h * 128;
  float x0 = row[lane], x1 = row[lane + 32];
  float p1 = x0 * ah[lane]      + x1 * ah[lane + 32];
  float p2 = x0 * ah[64 + lane] + x1 * ah[96 + lane];
#pragma unroll
  for (int o = 16; o > 0; o >>= 1) {
    p1 += __shfl_xor(p1, o, 32);
    p2 += __shfl_xor(p2, o, 32);
  }
  if (lane == 0) { s1[(size_t)h * N_NODES + n] = p1; s2[(size_t)h * N_NODES + n] = p2; }
}

// --------------------------------------------------------- softmax row stats
// Pass 1: row max m_i and 1/Z_i of exp(e_ij - m_i), e = masked leaky-relu.
__global__ __launch_bounds__(32)
void k_stats(const float* __restrict__ adj,
             const float* __restrict__ s1, const float* __restrict__ s2,
             float* __restrict__ mrow, float* __restrict__ invZ) {
  int i = blockIdx.x, h = blockIdx.y, lane = threadIdx.x;
  float si = s1[(size_t)h * N_NODES + i];
  const float* arow = adj + (size_t)i * N_NODES;
  const float* s2h  = s2 + (size_t)h * N_NODES;

  float m = NEG_INF_F;
  for (int j = lane; j < N_NODES; j += 32) {
    float t = si + s2h[j];
    float e = t > 0.f ? t : LRELU_ALPHA * t;
    e = (arow[j] > 0.f) ? e : NEG_INF_F;
    m = fmaxf(m, e);
  }
#pragma unroll
  for (int o = 16; o > 0; o >>= 1) m = fmaxf(m, __shfl_xor(m, o, 32));

  float z = 0.f;
  for (int j = lane; j < N_NODES; j += 32) {
    float t = si + s2h[j];
    float e = t > 0.f ? t : LRELU_ALPHA * t;
    float p = (arow[j] > 0.f) ? __expf(e - m) : 0.f;
    z += p;
  }
#pragma unroll
  for (int o = 16; o > 0; o >>= 1) z += __shfl_xor(z, o, 32);

  if (lane == 0) {
    mrow[(size_t)h * N_NODES + i] = m;
    invZ[(size_t)h * N_NODES + i] = 1.f / z;
  }
}

// ------------------------------------------------- fused attention apply P@V
// Pass 2: regenerate normalized P tiles (16 rows x 32 cols, f16) in a
// WMMA A layout (contiguous k = 16*hi + s, as in B) and multiply against V^T tiles (contiguous 32B per lane).
__global__ __launch_bounds__(32)
void k_att_apply(const float* __restrict__ adj,
                 const float* __restrict__ s1, const float* __restrict__ s2,
                 const float* __restrict__ mrow, const float* __restrict__ invZ,
                 const _Float16* __restrict__ vT,   // [H][64][N]
                 float* __restrict__ out32,         // [N][64] or nullptr
                 _Float16* __restrict__ out16,      // [N][stride] or nullptr
                 int out_stride, int do_elu) {
  const int h       = blockIdx.y;
  const int rowbase = blockIdx.x * 16;
  const int lane    = threadIdx.x;
  const int ml      = lane & 15;
  const int hi      = lane >> 4;
  const int irow    = rowbase + ml;

  const float si = s1[(size_t)h * N_NODES + irow];
  const float mi = mrow[(size_t)h * N_NODES + irow];
  const float iz = invZ[(size_t)h * N_NODES + irow];
  const float* arow = adj + (size_t)irow * N_NODES;
  const float* s2h  = s2 + (size_t)h * N_NODES;
  const _Float16* vTh = vT + (size_t)h * 64 * N_NODES;

  v8f c0 = {}, c1 = {}, c2 = {}, c3 = {};
  for (int jt = 0; jt < N_NODES; jt += 32) {
    if (jt + 128 < N_NODES)
      __builtin_prefetch(arow + jt + 128, 0, 1);   // -> global_prefetch_b8

    v16h a;
#pragma unroll
    for (int s = 0; s < 16; ++s) {
      int k = hi * 16 + s;
      int j = jt + k;
      float t = si + s2h[j];
      float e = t > 0.f ? t : LRELU_ALPHA * t;
      float p = (arow[j] > 0.f) ? __expf(e - mi) * iz : 0.f;
      a[s] = (_Float16)p;
    }

    v16h b0 = *(const v16h*)(vTh + (size_t)( 0 + ml) * N_NODES + jt + hi * 16);
    v16h b1 = *(const v16h*)(vTh + (size_t)(16 + ml) * N_NODES + jt + hi * 16);
    v16h b2 = *(const v16h*)(vTh + (size_t)(32 + ml) * N_NODES + jt + hi * 16);
    v16h b3 = *(const v16h*)(vTh + (size_t)(48 + ml) * N_NODES + jt + hi * 16);

    c0 = __builtin_amdgcn_wmma_f32_16x16x32_f16(false, a, false, b0, (short)0, c0, false, false);
    c1 = __builtin_amdgcn_wmma_f32_16x16x32_f16(false, a, false, b1, (short)0, c1, false, false);
    c2 = __builtin_amdgcn_wmma_f32_16x16x32_f16(false, a, false, b2, (short)0, c2, false, false);
    c3 = __builtin_amdgcn_wmma_f32_16x16x32_f16(false, a, false, b3, (short)0, c3, false, false);
  }

#pragma unroll
  for (int i = 0; i < 8; ++i) {
    int grow = rowbase + i + hi * 8;
    float v[4] = {c0[i], c1[i], c2[i], c3[i]};
#pragma unroll
    for (int t = 0; t < 4; ++t) {
      int col = t * 16 + ml;
      float val = v[t];
      if (do_elu) val = (val > 0.f) ? val : (__expf(val) - 1.f);
      if (out32) out32[(size_t)grow * 64 + col] = val;
      if (out16) out16[(size_t)grow * out_stride + h * 64 + col] = (_Float16)val;
    }
  }
}

// ---------------------------------------------------------------------------
extern "C" void kernel_launch(void* const* d_in, const int* in_sizes, int n_in,
                              void* d_out, int out_size, void* d_ws, size_t ws_size,
                              hipStream_t stream) {
  (void)in_sizes; (void)n_in; (void)out_size; (void)ws_size;
  const float* features = (const float*)d_in[0];  // [4096,256]
  const float* adj      = (const float*)d_in[1];  // [4096,4096]
  const float* W_heads  = (const float*)d_in[2];  // [8,256,64]
  const float* a_heads  = (const float*)d_in[3];  // [8,128,1]
  const float* W_out    = (const float*)d_in[4];  // [512,64]
  const float* a_out    = (const float*)d_in[5];  // [128,1]
  float* out = (float*)d_out;                     // [4096,64]

  const int N = N_NODES, I = 256, D = 64, H = 8, K2 = 512;

  // ---- workspace carve (256B-aligned chunks) ----
  char* w = (char*)d_ws;
  auto carve = [&](size_t bytes) -> char* {
    char* p = w; w += (bytes + 255) & ~(size_t)255; return p;
  };
  _Float16* fh    = (_Float16*)carve((size_t)N * I * 2);        // features f16
  _Float16* WhT   = (_Float16*)carve((size_t)H * D * I * 2);    // W1^T f16
  _Float16* WoT   = (_Float16*)carve((size_t)D * K2 * 2);       // W2^T f16
  float*    h1    = (float*)   carve((size_t)H * N * D * 4);    // layer1 h
  _Float16* h1T   = (_Float16*)carve((size_t)H * D * N * 2);    // layer1 h^T f16
  float*    s1_1  = (float*)   carve((size_t)H * N * 4);
  float*    s2_1  = (float*)   carve((size_t)H * N * 4);
  float*    m1    = (float*)   carve((size_t)H * N * 4);
  float*    iz1   = (float*)   carve((size_t)H * N * 4);
  _Float16* x2h   = (_Float16*)carve((size_t)N * K2 * 2);       // ELU-packed f16
  float*    h2    = (float*)   carve((size_t)N * D * 4);
  _Float16* h2T   = (_Float16*)carve((size_t)D * N * 2);
  float*    s1_2  = (float*)   carve((size_t)N * 4);
  float*    s2_2  = (float*)   carve((size_t)N * 4);
  float*    m2    = (float*)   carve((size_t)N * 4);
  float*    iz2   = (float*)   carve((size_t)N * 4);

  // ---- 1) precision conversion / transposes ----
  {
    int n = N * I;
    k_f32_to_f16<<<(n + 255) / 256, 256, 0, stream>>>(features, fh, n);
  }
  {
    int total = H * I * D;
    k_transpose_f16<<<(total + 255) / 256, 256, 0, stream>>>(W_heads, WhT, I, D, total);
  }
  {
    int total = K2 * D;
    k_transpose_f16<<<(total + 255) / 256, 256, 0, stream>>>(W_out, WoT, K2, D, total);
  }

  // ---- 2) layer-1 feature GEMM: h1 = x @ W (per head) ----
  k_gemm16<<<dim3(N / 16, H), 32, 0, stream>>>(fh, WhT, I, h1, h1T);

  // ---- 3) layer-1 attention scores ----
  k_scores<<<dim3(N, H), 32, 0, stream>>>(h1, a_heads, s1_1, s2_1);

  // ---- 4) layer-1 masked-softmax stats ----
  k_stats<<<dim3(N, H), 32, 0, stream>>>(adj, s1_1, s2_1, m1, iz1);

  // ---- 5) layer-1 fused attention + ELU + head-pack -> x2h [N, H*64] f16 ----
  k_att_apply<<<dim3(N / 16, H), 32, 0, stream>>>(
      adj, s1_1, s2_1, m1, iz1, h1T,
      (float*)nullptr, x2h, /*out_stride=*/K2, /*do_elu=*/1);

  // ---- 6) layer-2 GEMM: h2 = x2 @ W_out ----
  k_gemm16<<<dim3(N / 16, 1), 32, 0, stream>>>(x2h, WoT, K2, h2, h2T);

  // ---- 7) layer-2 scores / stats / attention -> d_out ----
  k_scores<<<dim3(N, 1), 32, 0, stream>>>(h2, a_out, s1_2, s2_2);
  k_stats<<<dim3(N, 1), 32, 0, stream>>>(adj, s1_2, s2_2, m2, iz2);
  k_att_apply<<<dim3(N / 16, 1), 32, 0, stream>>>(
      adj, s1_2, s2_2, m2, iz2, h2T,
      out, (_Float16*)nullptr, /*out_stride=*/64, /*do_elu=*/0);
}
